// CustomGCNConv_37254546325944
// MI455X (gfx1250) — matrix operand order repaired and checked by reading, not compiled, not run on hardware
//
#include <hip/hip_runtime.h>

#define NNODES 8192
#define NEDGES 262144
#define CH 64
#define WORDS_PER_ROW 256   // 8192 bits / 32

typedef __attribute__((ext_vector_type(16))) _Float16 v16h;
typedef __attribute__((ext_vector_type(8)))  float    v8f;

// ---------------------------------------------------------------- zero bitmap
__global__ __launch_bounds__(256) void k_zero(unsigned int* __restrict__ bm) {
    bm[(size_t)blockIdx.x * blockDim.x + threadIdx.x] = 0u;
}

// ------------------------------------------------- scatter edges into bitmap
// set-semantics: duplicate edges collapse because the bit is just re-OR'd
__global__ __launch_bounds__(256) void k_scatter(const int* __restrict__ ei,
                                                 unsigned int* __restrict__ bm) {
    int e = blockIdx.x * blockDim.x + threadIdx.x;
    int s = ei[e];             // edge_index[0][e]
    int d = ei[NEDGES + e];    // edge_index[1][e]
    atomicOr(&bm[(size_t)s * WORDS_PER_ROW + (d >> 5)], 1u << (d & 31));
}

// ---------------------------- degree via popcount, dinv, and y = dinv[j]*x[j]
// one wave32 per row; 64 channels -> one float2 per lane
__global__ __launch_bounds__(32) void k_degree(const unsigned int* __restrict__ bm,
                                               const float* __restrict__ x,
                                               float* __restrict__ dinv,
                                               float* __restrict__ y) {
    const int row  = blockIdx.x;          // uniform per wave -> scalar loads
    const int lane = threadIdx.x;
    const unsigned int* r = bm + (size_t)row * WORDS_PER_ROW;
    int pc = 0;
#pragma unroll
    for (int i = 0; i < 8; ++i) pc += __popc(r[lane * 8 + i]);
#pragma unroll
    for (int off = 16; off > 0; off >>= 1) pc += __shfl_xor(pc, off, 32);
    // degree = row-sum of (A_set + I) = popcount + 1 (self-loop bit adds 2 total)
    float di = rsqrtf((float)(pc + 1));
    if (lane == 0) dinv[row] = di;
    const float2* x2 = (const float2*)(x + (size_t)row * CH);
    float2*       y2 = (float2*)(y + (size_t)row * CH);
    float2 v = x2[lane];
    y2[lane] = make_float2(di * v.x, di * v.y);
}

// ------------------------------------- h[i] = dinv[i] * (y[i] + sum_nbr y[j])
// one wave32 per row; scan the row bitmap (uniform addresses -> scalar path)
__global__ __launch_bounds__(32) void k_aggregate(const unsigned int* __restrict__ bm,
                                                  const float* __restrict__ dinv,
                                                  const float* __restrict__ y,
                                                  float* __restrict__ h) {
    const int row  = blockIdx.x;
    const int lane = threadIdx.x;
    const unsigned int* r = bm + (size_t)row * WORDS_PER_ROW;
    const float2* y2 = (const float2*)y;
    float2 acc = y2[(size_t)row * 32 + lane];   // identity (eye) contribution
    for (int w = 0; w < WORDS_PER_ROW; ++w) {
        unsigned int word = r[w];
        while (word) {
            int j = (w << 5) + __builtin_ctz(word);
            word &= word - 1;
            float2 v = y2[(size_t)j * 32 + lane];
            acc.x += v.x;
            acc.y += v.y;
        }
    }
    float di = dinv[row];
    float2* h2 = (float2*)h;
    h2[(size_t)row * 32 + lane] = make_float2(di * acc.x, di * acc.y);
}

// ------------------------------------------------- out = h @ W^T + b via WMMA
// one wave32 per 16x16 output tile; K=64 as 2x v_wmma_f32_16x16x32_f16
__global__ __launch_bounds__(256) void k_gemm_wmma(const float* __restrict__ h,
                                                   const float* __restrict__ W,
                                                   const float* __restrict__ bias,
                                                   float* __restrict__ out) {
    const int lane = threadIdx.x & 31;
    const int wave = threadIdx.x >> 5;
    const int tile = blockIdx.x * 8 + wave;   // 2048 tiles total
    const int mt   = tile >> 2;               // 512 row tiles
    const int nt   = tile & 3;                // 4 col tiles
    const int half = lane >> 4;
    const int l16  = lane & 15;
    const int m = mt * 16 + l16;              // A-matrix row owned by this lane
    const int n = nt * 16 + l16;              // B/C/D column owned by this lane

    // C/D layout: VGPR v -> M = v + 8*half, N = l16 ; bias depends only on N
    v8f c;
    float bv = bias[n];
#pragma unroll
    for (int v = 0; v < 8; ++v) c[v] = bv;

#pragma unroll
    for (int k0 = 0; k0 < CH; k0 += 32) {
        v16h a, b;
        // A 16x32 f16: VGPR p<4 -> K = 2p + 8*half ; p>=4 -> K = 16 + 2(p-4) + 8*half
#pragma unroll
        for (int p = 0; p < 8; ++p) {
            int kb = ((p < 4) ? (2 * p) : (16 + 2 * (p - 4))) + 8 * half + k0;
            a[2 * p]     = (_Float16)h[(size_t)m * CH + kb];
            a[2 * p + 1] = (_Float16)h[(size_t)m * CH + kb + 1];
        }
        // B 32x16 f16 (col n): VGPR p<4 -> K = 2p + 8*half ; p>=4 -> K = 16 + 2(p-4) + 8*half ; B[k][n] = (W^T)[k][n] = W[n][k]
#pragma unroll
        for (int p = 0; p < 8; ++p) {
            int kb = ((p < 4) ? (2 * p) : (16 + 2 * (p - 4))) + 8 * half + k0;
            b[2 * p]     = (_Float16)W[(size_t)n * CH + kb];
            b[2 * p + 1] = (_Float16)W[(size_t)n * CH + kb + 1];
        }
        c = __builtin_amdgcn_wmma_f32_16x16x32_f16(false, a, false, b,
                                                   (short)0, c, false, false);
    }

#pragma unroll
    for (int v = 0; v < 8; ++v) {
        int mrow = mt * 16 + v + 8 * half;
        out[(size_t)mrow * CH + n] = c[v];
    }
}

// ---------------------------------------------------------------------------
extern "C" void kernel_launch(void* const* d_in, const int* in_sizes, int n_in,
                              void* d_out, int out_size, void* d_ws, size_t ws_size,
                              hipStream_t stream) {
    const float* x  = (const float*)d_in[0];   // [8192, 64]
    const int*   ei = (const int*)d_in[1];     // [2, 262144]
    const float* W  = (const float*)d_in[2];   // [64, 64]
    const float* b  = (const float*)d_in[3];   // [64]
    float* out = (float*)d_out;                // [8192, 64]

    unsigned char* ws   = (unsigned char*)d_ws;
    unsigned int*  bm   = (unsigned int*)ws;                                   // 8 MB bitmap
    float*         dinv = (float*)(ws + (size_t)NNODES * WORDS_PER_ROW * 4);   // 32 KB
    float*         y    = dinv + NNODES;                                       // 2 MB
    float*         h    = y + (size_t)NNODES * CH;                             // 2 MB

    k_zero     <<<NNODES * WORDS_PER_ROW / 256, 256, 0, stream>>>(bm);
    k_scatter  <<<NEDGES / 256,                 256, 0, stream>>>(ei, bm);
    k_degree   <<<NNODES,                        32, 0, stream>>>(bm, x, dinv, y);
    k_aggregate<<<NNODES,                        32, 0, stream>>>(bm, dinv, y, h);
    k_gemm_wmma<<<(NNODES / 16) * 4 / 8,        256, 0, stream>>>(h, W, b, out);
}
